// GAT_He_71116068488100
// MI455X (gfx1250) — compile-verified, matrix operand order checked by reading, not run on hardware
//
#include <hip/hip_runtime.h>
#include <hip/hip_fp16.h>

typedef __attribute__((ext_vector_type(16))) _Float16 v16h;
typedef __attribute__((ext_vector_type(8)))  _Float16 v8h;
typedef __attribute__((ext_vector_type(8)))  float    v8f;

#define N_NODES 30000
#define E_EDGES 600000
#define DN      128
#define HID     64
#define HEADS   4
#define ED      16
#define NC      10
#define NG      64
#define NEG_SLOPE 0.2f
#define TE (E_EDGES + N_NODES)   // edges + self loops

// ---------------------------------------------------------------------------
// f32 -> f16 convert (activations for WMMA A operand)
// ---------------------------------------------------------------------------
__global__ void cvt_f16_kernel(const float* __restrict__ s, _Float16* __restrict__ d,
                               long n) {
  long i = (long)blockIdx.x * blockDim.x + threadIdx.x;
  if (i < n) d[i] = (_Float16)s[i];
}

// ---------------------------------------------------------------------------
// Pack W [Ncols,K] f32 into f16 WMMA-fragment order:
//   frag(nt,kb): 32 lanes x 16 halves, contiguous (1 KB each).
// Per ISA 7.12.2 (16-bit 16x32): lane = (n&15) + 16*((k>>3)&1),
//   elem j = 8*((k&31)>>4) + (k&7).
// ---------------------------------------------------------------------------
__global__ void pack_w_kernel(const float* __restrict__ W, _Float16* __restrict__ Wp,
                              int K, int Ncols) {
  int i = blockIdx.x * blockDim.x + threadIdx.x;
  if (i >= Ncols * K) return;
  int n = i / K, k = i - n * K;
  int KB = K >> 5;
  int nt = n >> 4, kb = k >> 5;
  int lane = (n & 15) + (((k >> 3) & 1) << 4);
  int j = (((k & 31) >> 4) << 3) + (k & 7);
  Wp[(((size_t)(nt * KB + kb) * 32) + lane) * 16 + j] = (_Float16)W[(size_t)n * K + k];
}

// ---------------------------------------------------------------------------
// C[M,Ncols] = A16[M,K] @ W^T using packed W fragments.
// One wave computes a 16(M) x 64(N) strip: 1 A frag x 4 B frags -> 4 WMMA / K-step.
// M mult of 16, Ncols mult of 64, K mult of 32.
// ---------------------------------------------------------------------------
__global__ void gemm_wmma_kernel(const _Float16* __restrict__ A16,
                                 const _Float16* __restrict__ Wp,
                                 float* __restrict__ C,
                                 int M, int K, int Ncols) {
  int wave = blockIdx.x * (blockDim.x >> 5) + (threadIdx.x >> 5);
  int lane = threadIdx.x & 31;
  int KB = K >> 5;
  int nblk = Ncols >> 6;
  int T = (M >> 4) * nblk;
  if (wave >= T) wave = T - 1;            // tail waves redo a strip (same values)
  int mt = wave / nblk, nb = wave - mt * nblk;
  int m0 = mt << 4;

  size_t fragstride = (size_t)KB * 512;   // halves per packed n-tile
  const _Float16* bbase = Wp + ((size_t)(nb << 2) * KB * 32 + lane) * 16;
  const _Float16* arow  = A16 + (size_t)(m0 + (lane & 15)) * K + ((lane >> 4) & 1) * 8;

  v8f acc0 = {}, acc1 = {}, acc2 = {}, acc3 = {};
  for (int kb = 0; kb < KB; ++kb) {
    v8h alo = *(const v8h*)(arow + kb * 32);
    v8h ahi = *(const v8h*)(arow + kb * 32 + 16);
    v16h a = __builtin_shufflevector(alo, ahi, 0, 1, 2, 3, 4, 5, 6, 7,
                                     8, 9, 10, 11, 12, 13, 14, 15);
    const _Float16* bp = bbase + (size_t)kb * 512;
    v16h b0 = *(const v16h*)(bp);
    v16h b1 = *(const v16h*)(bp + fragstride);
    v16h b2 = *(const v16h*)(bp + 2 * fragstride);
    v16h b3 = *(const v16h*)(bp + 3 * fragstride);
    acc0 = __builtin_amdgcn_wmma_f32_16x16x32_f16(false, a, false, b0, (short)0, acc0, false, false);
    acc1 = __builtin_amdgcn_wmma_f32_16x16x32_f16(false, a, false, b1, (short)0, acc1, false, false);
    acc2 = __builtin_amdgcn_wmma_f32_16x16x32_f16(false, a, false, b2, (short)0, acc2, false, false);
    acc3 = __builtin_amdgcn_wmma_f32_16x16x32_f16(false, a, false, b3, (short)0, acc3, false, false);
  }
  // D layout: VGPR v -> M = v + 8*(lane>=16), N = lane&15
  int rowb = m0 + ((lane >> 4) & 1) * 8;
  int col0 = (nb << 6) + (lane & 15);
#pragma unroll
  for (int v = 0; v < 8; ++v) {
    float* crow = C + (size_t)(rowb + v) * Ncols + col0;
    crow[0]  = acc0[v];
    crow[16] = acc1[v];
    crow[32] = acc2[v];
    crow[48] = acc3[v];
  }
}

// ---------------------------------------------------------------------------
// Self-loop attr = per-dst mean of incoming edge_attr
// ---------------------------------------------------------------------------
__global__ void loop_sum_kernel(const int* __restrict__ dst,
                                const float* __restrict__ ea,
                                float* __restrict__ loop_attr,
                                float* __restrict__ cnt) {
  int e = blockIdx.x * blockDim.x + threadIdx.x;
  if (e >= E_EDGES) return;
  int d = dst[e];
  unsafeAtomicAdd(&cnt[d], 1.0f);
  const float* ep = ea + (size_t)e * ED;
  float* lp = loop_attr + (size_t)d * ED;
#pragma unroll
  for (int k = 0; k < ED; ++k) unsafeAtomicAdd(&lp[k], ep[k]);
}

__global__ void loop_div_kernel(float* __restrict__ loop_attr,
                                const float* __restrict__ cnt) {
  int i = blockIdx.x * blockDim.x + threadIdx.x;
  if (i >= N_NODES * ED) return;
  loop_attr[i] /= fmaxf(cnt[i / ED], 1.0f);
}

// ---------------------------------------------------------------------------
// Per-node attention terms: al_s[n,h] = h[n,h,:].a_s[h]; al_d likewise
// ---------------------------------------------------------------------------
__global__ void node_alpha_kernel(const float* __restrict__ h,
                                  const float* __restrict__ a_s,
                                  const float* __restrict__ a_d,
                                  float* __restrict__ als, float* __restrict__ ald,
                                  int heads) {
  int i = blockIdx.x * blockDim.x + threadIdx.x;
  if (i >= N_NODES * heads) return;
  int n = i / heads, hh = i - n * heads;
  const float* hp  = h + ((size_t)n * heads + hh) * HID;
  const float* asp = a_s + hh * HID;
  const float* adp = a_d + hh * HID;
  float s = 0.f, d = 0.f;
#pragma unroll 8
  for (int c = 0; c < HID; ++c) { float v = hp[c]; s += v * asp[c]; d += v * adp[c]; }
  als[i] = s; ald[i] = d;
}

// M[d,h] = sum_c We[h*HID+c, d] * ae[h, c]   (fused edge projection)
__global__ void edge_proj_kernel(const float* __restrict__ We,
                                 const float* __restrict__ ae,
                                 float* __restrict__ Mm, int heads) {
  int i = threadIdx.x;
  if (i >= ED * heads) return;
  int d = i / heads, hh = i - d * heads;
  float s = 0.f;
  for (int c = 0; c < HID; ++c)
    s += We[(size_t)(hh * HID + c) * ED + d] * ae[hh * HID + c];
  Mm[d * heads + hh] = s;
}

// alpha -> exp(alpha), accumulate softmax denominators per (dst, head).
// (max-subtraction cancels in softmax; magnitudes here are small enough for f32)
__global__ void edge_attn_kernel(const int* __restrict__ src, const int* __restrict__ dst,
                                 const float* __restrict__ ea,
                                 const float* __restrict__ loop_attr,
                                 const float* __restrict__ als, const float* __restrict__ ald,
                                 const float* __restrict__ Mm,
                                 float* __restrict__ ex, float* __restrict__ den,
                                 int heads) {
  int e = blockIdx.x * blockDim.x + threadIdx.x;
  if (e >= TE) return;
  int s, d; const float* ep;
  if (e < E_EDGES) { s = src[e]; d = dst[e]; ep = ea + (size_t)e * ED; }
  else             { s = e - E_EDGES; d = s; ep = loop_attr + (size_t)s * ED; }
  float eav[ED];
#pragma unroll
  for (int k = 0; k < ED; ++k) eav[k] = ep[k];
  for (int hh = 0; hh < heads; ++hh) {
    float ale = 0.f;
#pragma unroll
    for (int k = 0; k < ED; ++k) ale += eav[k] * Mm[k * heads + hh];
    float a = als[s * heads + hh] + ald[d * heads + hh] + ale;
    a = (a > 0.f) ? a : NEG_SLOPE * a;
    float x = __expf(a);
    ex[(size_t)e * heads + hh] = x;
    unsafeAtomicAdd(&den[d * heads + hh], x);
  }
}

// out[dst] += h[src] * att ; 64 threads per edge (one per channel)
__global__ void aggregate_kernel(const int* __restrict__ src, const int* __restrict__ dst,
                                 const float* __restrict__ h,
                                 const float* __restrict__ ex, const float* __restrict__ den,
                                 float* __restrict__ out, int heads) {
  int e = blockIdx.x * (blockDim.x >> 6) + (threadIdx.x >> 6);
  int t = threadIdx.x & 63;
  if (e >= TE) return;
  int s, d;
  if (e < E_EDGES) { s = src[e]; d = dst[e]; }
  else             { s = e - E_EDGES; d = s; }
  for (int hh = 0; hh < heads; ++hh) {
    float w = ex[(size_t)e * heads + hh] / (den[d * heads + hh] + 1e-16f);
    float v = h[((size_t)s * heads + hh) * HID + t] * w;
    unsafeAtomicAdd(&out[((size_t)d * heads + hh) * HID + t], v);
  }
}

// bias + relu, emits f32 (for pooling / validation) and f16 (next GEMM A operand)
__global__ void bias_relu_kernel(float* __restrict__ p, _Float16* __restrict__ p16,
                                 const float* __restrict__ b, int F) {
  size_t i = (size_t)blockIdx.x * blockDim.x + threadIdx.x;
  if (i >= (size_t)N_NODES * F) return;
  float v = p[i] + b[i % F];
  v = v > 0.f ? v : 0.f;
  p[i] = v;
  p16[i] = (_Float16)v;
}

// ---------------------------------------------------------------------------
// Global mean pool + classifier + softmax
// ---------------------------------------------------------------------------
__global__ void pool_kernel(const float* __restrict__ p, const int* __restrict__ batch,
                            float* __restrict__ pooled, float* __restrict__ gcnt) {
  int i = blockIdx.x * blockDim.x + threadIdx.x;
  if (i >= N_NODES * HID) return;
  int n = i / HID, c = i - n * HID;
  int g = batch[n];
  unsafeAtomicAdd(&pooled[g * HID + c], p[i]);
  if (c == 0) unsafeAtomicAdd(&gcnt[g], 1.0f);
}

__global__ void classify_kernel(const float* __restrict__ pooled, const float* __restrict__ gcnt,
                                const float* __restrict__ Wf, const float* __restrict__ bf,
                                float* __restrict__ out) {
  int g = threadIdx.x;
  if (g >= NG) return;
  float inv = 1.0f / fmaxf(gcnt[g], 1.0f);
  float feat[HID];
#pragma unroll
  for (int c = 0; c < HID; ++c) feat[c] = pooled[g * HID + c] * inv;
  float logit[NC]; float mx = -1e30f;
  for (int k = 0; k < NC; ++k) {
    float s = bf[k];
    for (int c = 0; c < HID; ++c) s += feat[c] * Wf[k * HID + c];
    logit[k] = s; mx = fmaxf(mx, s);
  }
  float sum = 0.f;
  for (int k = 0; k < NC; ++k) { logit[k] = __expf(logit[k] - mx); sum += logit[k]; }
  for (int k = 0; k < NC; ++k) out[g * NC + k] = logit[k] / sum;
}

// ---------------------------------------------------------------------------
extern "C" void kernel_launch(void* const* d_in, const int* in_sizes, int n_in,
                              void* d_out, int out_size, void* d_ws, size_t ws_size,
                              hipStream_t stream) {
  const float* x     = (const float*)d_in[0];
  const int*   ei    = (const int*)  d_in[1];
  const float* eattr = (const float*)d_in[2];
  const int*   batch = (const int*)  d_in[3];
  const float* W1  = (const float*)d_in[4];
  const float* as1 = (const float*)d_in[5];
  const float* ad1 = (const float*)d_in[6];
  const float* We1 = (const float*)d_in[7];
  const float* ae1 = (const float*)d_in[8];
  const float* b1  = (const float*)d_in[9];
  const float* W2  = (const float*)d_in[10];
  const float* as2 = (const float*)d_in[11];
  const float* ad2 = (const float*)d_in[12];
  const float* We2 = (const float*)d_in[13];
  const float* ae2 = (const float*)d_in[14];
  const float* b2  = (const float*)d_in[15];
  const float* W3  = (const float*)d_in[16];
  const float* as3 = (const float*)d_in[17];
  const float* ad3 = (const float*)d_in[18];
  const float* We3 = (const float*)d_in[19];
  const float* ae3 = (const float*)d_in[20];
  const float* b3  = (const float*)d_in[21];
  const float* Wf  = (const float*)d_in[22];
  const float* bf  = (const float*)d_in[23];

  const int* src = ei;            // edge_index[0]
  const int* dst = ei + E_EDGES;  // edge_index[1]

  // workspace layout (float region, then f16 region; all offsets 64B aligned)
  float* ws = (float*)d_ws;
  size_t off = 0;
  float* loop_attr = ws + off; off += (size_t)N_NODES * ED;
  float* cnt       = ws + off; off += N_NODES;
  float* hbuf      = ws + off; off += (size_t)N_NODES * HEADS * HID;
  float* p0        = ws + off; off += (size_t)N_NODES * HEADS * HID;
  float* p1        = ws + off; off += (size_t)N_NODES * HEADS * HID;
  float* als       = ws + off; off += (size_t)N_NODES * HEADS;
  float* ald       = ws + off; off += (size_t)N_NODES * HEADS;
  float* Mm        = ws + off; off += 64;  // ED*HEADS padded
  float* exb       = ws + off; off += (size_t)TE * HEADS;
  float* den       = ws + off; off += (size_t)N_NODES * HEADS;
  float* pooled    = ws + off; off += NG * HID;
  float* gcnt      = ws + off; off += NG;
  _Float16* act16  = (_Float16*)(ws + off);            // N x 256 halves
  _Float16* wp1    = act16 + (size_t)N_NODES * 256;    // 256x128
  _Float16* wp2    = wp1 + 256 * 128;                  // 256x256
  _Float16* wp3    = wp2 + 256 * 256;                  // 64x256

  // ---- pack weights into WMMA fragment order (f16) ----
  pack_w_kernel<<<(256 * 128 + 255) / 256, 256, 0, stream>>>(W1, wp1, DN, 256);
  pack_w_kernel<<<(256 * 256 + 255) / 256, 256, 0, stream>>>(W2, wp2, 256, 256);
  pack_w_kernel<<<(64 * 256 + 255) / 256, 256, 0, stream>>>(W3, wp3, 256, 64);

  // ---- self-loop edge attr (mean of incoming) ----
  hipMemsetAsync(loop_attr, 0, (size_t)(N_NODES * ED + N_NODES) * sizeof(float), stream);
  loop_sum_kernel<<<(E_EDGES + 255) / 256, 256, 0, stream>>>(dst, eattr, loop_attr, cnt);
  loop_div_kernel<<<(N_NODES * ED + 255) / 256, 256, 0, stream>>>(loop_attr, cnt);

  // ---- x -> f16 for layer-1 A operand ----
  cvt_f16_kernel<<<(int)(((long)N_NODES * DN + 255) / 256), 256, 0, stream>>>(
      x, act16, (long)N_NODES * DN);

  auto run_layer = [&](int K, const _Float16* Wp,
                       const float* a_s, const float* a_d,
                       const float* We, const float* a_e, const float* b,
                       int heads, float* pout) {
    int Ncols = heads * HID;
    int T = (N_NODES / 16) * (Ncols / 64);
    gemm_wmma_kernel<<<(T + 7) / 8, 256, 0, stream>>>(act16, Wp, hbuf, N_NODES, K, Ncols);
    node_alpha_kernel<<<(N_NODES * heads + 255) / 256, 256, 0, stream>>>(
        hbuf, a_s, a_d, als, ald, heads);
    edge_proj_kernel<<<1, 64, 0, stream>>>(We, a_e, Mm, heads);
    hipMemsetAsync(den, 0, (size_t)N_NODES * heads * sizeof(float), stream);
    edge_attn_kernel<<<(TE + 255) / 256, 256, 0, stream>>>(
        src, dst, eattr, loop_attr, als, ald, Mm, exb, den, heads);
    hipMemsetAsync(pout, 0, (size_t)N_NODES * Ncols * sizeof(float), stream);
    aggregate_kernel<<<(TE + 3) / 4, 256, 0, stream>>>(src, dst, hbuf, exb, den, pout, heads);
    // overwrites act16 with this layer's output (previous GEMM read already done)
    bias_relu_kernel<<<(int)(((size_t)N_NODES * Ncols + 255) / 256), 256, 0, stream>>>(
        pout, act16, b, Ncols);
  };

  run_layer(DN,          wp1, as1, ad1, We1, ae1, b1, HEADS, p0);
  run_layer(HEADS * HID, wp2, as2, ad2, We2, ae2, b2, HEADS, p1);
  run_layer(HEADS * HID, wp3, as3, ad3, We3, ae3, b3, 1,     p0);

  // ---- pool + classify ----
  hipMemsetAsync(pooled, 0, (size_t)(NG * HID + NG) * sizeof(float), stream);
  pool_kernel<<<(N_NODES * HID + 255) / 256, 256, 0, stream>>>(p0, batch, pooled, gcnt);
  classify_kernel<<<1, 64, 0, stream>>>(pooled, gcnt, Wf, bf, (float*)d_out);
}
